// FlowMatchingHead_51196010168967
// MI455X (gfx1250) — matrix operand order repaired and checked by reading, not compiled, not run on hardware
//
#include <hip/hip_runtime.h>
#include <hip/hip_bf16.h>

// ---------------------------------------------------------------------------
// FlowMatchingHead on MI455X (gfx1250, wave32, WMMA + async LDS DMA)
//   E=32, A=32, H=512, B=128, T=64  -> BT = 8192 tokens
//   out = CatLin3( swish( CatLin2( [CatLin1(actions) | tau_emb] ) ) )
// bf16 WMMA (16x16x32) with f32 accumulate; fp32 weights converted to bf16
// while staging through LDS (transposed tile -> contiguous B fragments).
// bf16 activation tiles are DMA'd global->LDS with
// GLOBAL_LOAD_ASYNC_TO_LDS_B128 (ASYNCcnt), overlapped with weight staging.
// ---------------------------------------------------------------------------

typedef __attribute__((ext_vector_type(16))) __bf16 v16bf;
typedef __attribute__((ext_vector_type(8)))  __bf16 v8bf;
typedef __attribute__((ext_vector_type(4)))  __bf16 v4bf;
typedef __attribute__((ext_vector_type(8)))  float  v8f;
typedef __attribute__((ext_vector_type(4)))  float  v4f;

#define E_ 32
#define A_ 32
#define H_ 512
#define B_ 128
#define T_ 64

#define TB 128        // 4 waves per workgroup
#define KC 64         // K chunk staged through LDS
#define KP (KC + 16)  // padded row (160B stride, keeps 16B/8B alignment)

// -------------------------------------------------------------------
// Kernel 1: sinusoidal timestep embedding -> bf16 into concat buffer
// Xc layout: (B*T, 1024) bf16 ; tau occupies columns 512..1023
// -------------------------------------------------------------------
__global__ __launch_bounds__(256) void tau_embed_kernel(
    const int* __restrict__ timesteps, __bf16* __restrict__ Xc)
{
    const int tok = blockIdx.x;            // 0 .. B*T-1
    const int j   = threadIdx.x;           // 0 .. 255   (half = 256)
    const float t = (float)timesteps[tok];
    const float kf = -9.210340371976184f / 256.0f;   // -ln(10000)/256
    const float f  = t * __expf(kf * (float)j);
    __bf16* row = Xc + (size_t)tok * 1024 + 512;
    row[j]       = (__bf16)__sinf(f);
    row[j + 256] = (__bf16)__cosf(f);
}

// Combine two contiguous v8bf LDS loads into one v16bf fragment.
__device__ __forceinline__ v16bf load_frag(const __bf16* p0, const __bf16* p1)
{
    v8bf lo = *(const v8bf*)p0;
    v8bf hi = *(const v8bf*)p1;
    return __builtin_shufflevector(lo, hi,
        0, 1, 2, 3, 4, 5, 6, 7, 8, 9, 10, 11, 12, 13, 14, 15);
}

// -------------------------------------------------------------------
// Generic category-gathered GEMM:
//   Out[b, :, cb*64 : cb*64+64] = act( X[b](T_ x K) @ W[cat(b)](K x 512) + bias )
// Grid: (N/64, B).  One workgroup = 64x64 output tile, 4 waves,
// wave w computes rows [16w,16w+16) x all 64 cols (4 WMMA col tiles).
// -------------------------------------------------------------------
template<bool XF32, bool SWISH, bool OUTBF>
__global__ __launch_bounds__(TB) void gemm_cat_kernel(
    const void*  __restrict__ Xbase, int ldx,
    const float* __restrict__ Wbase,       // (E, K, 512) fp32
    const float* __restrict__ Bias,        // (E, 512)    fp32
    int K,
    const int*   __restrict__ cat_ids,
    void*        __restrict__ Outbase, int ldo)
{
    __shared__ __attribute__((aligned(16))) __bf16 Xs [64][KP]; // act tile [row][k]
    __shared__ __attribute__((aligned(16))) __bf16 WsT[64][KP]; // wgt tile [col][k]

    const int b    = blockIdx.y;
    const int cb   = blockIdx.x;          // 64-column block
    const int cat  = cat_ids[b];
    const int tid  = threadIdx.x;
    const int lane = tid & 31;
    const int wave = tid >> 5;            // row tile 0..3

    const float* W    = Wbase + (size_t)cat * K * H_ + (size_t)cb * 64;
    const float* bias = Bias  + (size_t)cat * H_     + (size_t)cb * 64;

    const float*  Xf = (const float*) Xbase + (size_t)b * T_ * ldx;
    const __bf16* Xh = (const __bf16*)Xbase + (size_t)b * T_ * ldx;

    v8f acc[4];
    #pragma unroll
    for (int ct = 0; ct < 4; ++ct)
        #pragma unroll
        for (int r = 0; r < 8; ++r) acc[ct][r] = 0.0f;

    // X staging roles: thread -> (row = tid>>1, 32-wide half = tid&1)
    const int srow = tid >> 1;            // 0..63
    const int sc0  = (tid & 1) * 32;      // 0 or 32
    // W staging roles: thread -> 4(k) x 8(col) block
    const int wk4  = (tid >> 3) << 2;     // k base 0,4,...,60
    const int wcg  = (tid & 7) << 3;      // col base 0,8,...,56

    // per-lane LDS byte offset of this thread's X destination
    // (generic LDS address: low 32 bits are the LDS byte offset)
    const unsigned xlds = (unsigned)(uintptr_t)&Xs[srow][sc0];

    for (int k0 = 0; k0 < K; k0 += KC) {
        // ---- stage X tile: Xs[row][k] (bf16) ----
        const bool okx = (k0 + sc0) < K;  // uniform per 32-group (K % 32 == 0)
        if (XF32) {
            #pragma unroll
            for (int i = 0; i < 4; ++i) {
                v8bf o;
                if (okx) {
                    const float* p = &Xf[(size_t)srow * ldx + k0 + sc0 + i * 8];
                    v4f f0 = *(const v4f*)p;
                    v4f f1 = *(const v4f*)(p + 4);
                    #pragma unroll
                    for (int e = 0; e < 4; ++e) {
                        o[e]     = (__bf16)f0[e];
                        o[4 + e] = (__bf16)f1[e];
                    }
                } else {
                    #pragma unroll
                    for (int e = 0; e < 8; ++e) o[e] = (__bf16)0.0f;
                }
                *(v8bf*)&Xs[srow][sc0 + i * 8] = o;
            }
        } else {
            // gfx1250 async DMA: global bf16 -> LDS, bypassing VGPRs.
            // Each lane copies a 64B half-row as 4 x b128 segments;
            // INST_OFFSET applies to both global and LDS addresses.
            const __bf16* gp = &Xh[(size_t)srow * ldx + k0 + sc0];
            asm volatile("global_load_async_to_lds_b128 %0, %1, off"
                         :: "v"(xlds), "v"(gp) : "memory");
            asm volatile("global_load_async_to_lds_b128 %0, %1, off offset:16"
                         :: "v"(xlds), "v"(gp) : "memory");
            asm volatile("global_load_async_to_lds_b128 %0, %1, off offset:32"
                         :: "v"(xlds), "v"(gp) : "memory");
            asm volatile("global_load_async_to_lds_b128 %0, %1, off offset:48"
                         :: "v"(xlds), "v"(gp) : "memory");
        }

        // ---- stage W tile transposed: WsT[col][k] (bf16), b64 stores ----
        // (overlaps with the in-flight async X DMA)
        {
            const bool okw = (k0 + wk4) < K;     // uniform (K % 32 == 0)
            v4f fr[4][2];                        // 4 rows x 8 cols
            #pragma unroll
            for (int r = 0; r < 4; ++r) {
                if (okw) {
                    const float* p = &W[(size_t)(k0 + wk4 + r) * H_ + wcg];
                    fr[r][0] = *(const v4f*)p;
                    fr[r][1] = *(const v4f*)(p + 4);
                } else {
                    #pragma unroll
                    for (int e = 0; e < 4; ++e) { fr[r][0][e] = 0.0f; fr[r][1][e] = 0.0f; }
                }
            }
            if ((k0 + KC) < K)                   // prefetch next weight chunk
                __builtin_prefetch(&W[(size_t)(k0 + KC + wk4) * H_ + wcg], 0, 3);
            #pragma unroll
            for (int j = 0; j < 8; ++j) {
                v4bf o;
                #pragma unroll
                for (int r = 0; r < 4; ++r) o[r] = (__bf16)fr[r][j >> 2][j & 3];
                *(v4bf*)&WsT[wcg + j][wk4] = o;  // 8B: 4 bf16 along k
            }
        }

        if (!XF32)   // drain this wave's async LDS writes before the barrier
            asm volatile("s_wait_asynccnt 0x0" ::: "memory");
        __syncthreads();

        // ---- two K=32 WMMA steps over the 64-deep chunk ----
        #pragma unroll
        for (int kk = 0; kk < KC; kk += 32) {
            // A fragment (16x32): lane = row m; K-halves split by lane>>4
            //   a[0..7] = Xs[m][kk+kh .. +7], a[8..15] = Xs[m][kk+kh+16 .. +23]
            const int m  = (wave << 4) + (lane & 15);
            const int kh = (lane >> 4) << 3;        // 0 or 8
            const __bf16* ap = &Xs[m][kk + kh];
            v16bf a = load_frag(ap, ap + 16);

            #pragma unroll
            for (int ct = 0; ct < 4; ++ct) {
                // B fragment (32x16): same K split as A
                //   b[0..7] = WsT[n][kb .. +7], b[8..15] = WsT[n][kb+16 .. +23]
                const int n  = (ct << 4) + (lane & 15);
                const int kb = kk + ((lane >> 4) << 3);
                const __bf16* bp = &WsT[n][kb];
                v16bf bfr = load_frag(bp, bp + 16);

                acc[ct] = __builtin_amdgcn_wmma_f32_16x16x32_bf16(
                    false, a, false, bfr, (short)0, acc[ct], false, false);
            }
        }
        __syncthreads();
    }

    // ---- epilogue: bias (+ swish), store ----
    // D layout: VGPR r -> row = r + 8*(lane>>4); col = lane&15
    #pragma unroll
    for (int ct = 0; ct < 4; ++ct) {
        const int coln = (ct << 4) + (lane & 15);
        const float bv = bias[coln];
        const int rbase = (wave << 4) + ((lane >> 4) << 3);
        #pragma unroll
        for (int r = 0; r < 8; ++r) {
            float v = acc[ct][r] + bv;
            if (SWISH) v = v / (1.0f + __expf(-v));
            const size_t oidx =
                ((size_t)b * T_ + (rbase + r)) * ldo + (size_t)cb * 64 + coln;
            if (OUTBF) ((__bf16*)Outbase)[oidx] = (__bf16)v;
            else       ((float*) Outbase)[oidx] = v;
        }
    }
}

// -------------------------------------------------------------------
// Launch: tau embed -> L1 (fp32 actions, K=32) -> L2 (K=1024, swish)
//         -> L3 (K=512, fp32 out)
// ws: Xc (B*T,1024) bf16 = 16 MB ; Hs (B*T,512) bf16 = 8 MB
// -------------------------------------------------------------------
extern "C" void kernel_launch(void* const* d_in, const int* in_sizes, int n_in,
                              void* d_out, int out_size, void* d_ws, size_t ws_size,
                              hipStream_t stream) {
    const float* actions   = (const float*)d_in[0];   // (B,T,32)
    const int*   timesteps = (const int*)  d_in[1];   // (B,T)
    const int*   cat_ids   = (const int*)  d_in[2];   // (B,)
    const float* W1        = (const float*)d_in[3];   // (E,32,512)
    const float* b1        = (const float*)d_in[4];   // (E,512)
    const float* W2        = (const float*)d_in[5];   // (E,1024,512)
    const float* b2        = (const float*)d_in[6];   // (E,512)
    const float* W3        = (const float*)d_in[7];   // (E,512,512)
    const float* b3        = (const float*)d_in[8];   // (E,512)
    float*       out       = (float*)d_out;           // (B,T,512)

    __bf16* Xc = (__bf16*)d_ws;                       // concat buffer (B*T,1024)
    __bf16* Hs = Xc + (size_t)B_ * T_ * 1024;         // hidden buffer (B*T,512)

    // tau embedding into columns 512..1023 of Xc
    tau_embed_kernel<<<B_ * T_, 256, 0, stream>>>(timesteps, Xc);

    // layer 1: actions(f32, ld=32) @ W1 -> Xc cols 0..511 (bf16, ld=1024)
    gemm_cat_kernel<true, false, true>
        <<<dim3(H_ / 64, B_), TB, 0, stream>>>(actions, A_, W1, b1, A_,
                                               cat_ids, Xc, 1024);

    // layer 2: Xc(bf16, ld=1024, K=1024) @ W2 -> swish -> Hs (bf16, ld=512)
    gemm_cat_kernel<false, true, true>
        <<<dim3(H_ / 64, B_), TB, 0, stream>>>(Xc, 1024, W2, b2, 2 * H_,
                                               cat_ids, Hs, H_);

    // layer 3: Hs(bf16, ld=512, K=512) @ W3 -> out (fp32, ld=512)
    gemm_cat_kernel<false, false, false>
        <<<dim3(H_ / 64, B_), TB, 0, stream>>>(Hs, H_, W3, b3, H_,
                                               cat_ids, out, H_);
}
